// qNetwork_SNN_21586505629720
// MI455X (gfx1250) — matrix operand order repaired and checked by reading, not compiled, not run on hardware
//
#include <hip/hip_runtime.h>
#include <stdint.h>

typedef __attribute__((ext_vector_type(16))) _Float16 v16h;
typedef __attribute__((ext_vector_type(8)))  _Float16 v8h;
typedef __attribute__((ext_vector_type(8)))  float    v8f;
typedef __attribute__((ext_vector_type(4)))  int      v4i;

#define TSTEPS 16
#define BETA 0.95f

#define BM 128
#define BN 128
#define BK 32
#define LDA 40   // (BK + 8) f16: 80B row stride, 16B aligned, pad kills bank conflicts
#define LDB 40

#if defined(__has_builtin)
#  if __has_builtin(__builtin_amdgcn_global_load_async_to_lds_b128)
#    define HAVE_ASYNC_BUILTIN 1
#  endif
#  if __has_builtin(__builtin_amdgcn_s_wait_asynccnt)
#    define HAVE_WAIT_BUILTIN 1
#  endif
#endif

// AS-qualified int4 pointee types (clang prints AS1 as __device__, AS3 as __shared__)
typedef __attribute__((address_space(1))) v4i as1_v4i;
typedef __attribute__((address_space(3))) v4i as3_v4i;

// Async copy 16B global -> LDS (ASYNCcnt-tracked, no VGPR roundtrip)
__device__ __forceinline__ void async_copy16(const _Float16* g, _Float16* l) {
#if defined(HAVE_ASYNC_BUILTIN)
    __builtin_amdgcn_global_load_async_to_lds_b128(
        (as1_v4i*)(uintptr_t)g,
        (as3_v4i*)(uint32_t)(uintptr_t)l,   // low 32 bits of flat LDS addr == LDS offset
        0, 0);
#else
    unsigned loff = (uint32_t)(uintptr_t)l;
    asm volatile("global_load_async_to_lds_b128 %0, %1, off"
                 :: "v"(loff), "v"(g) : "memory");
#endif
}

__device__ __forceinline__ void wait_async_le(int n) {
#if defined(HAVE_WAIT_BUILTIN)
    if (n == 4) __builtin_amdgcn_s_wait_asynccnt((unsigned short)4);
    else        __builtin_amdgcn_s_wait_asynccnt((unsigned short)0);
#else
    if (n == 4) asm volatile("s_wait_asynccnt 0x4" ::: "memory");
    else        asm volatile("s_wait_asynccnt 0x0" ::: "memory");
#endif
}

__device__ __forceinline__ v16h combine8(v8h lo, v8h hi) {
    v16h r;
#pragma unroll
    for (int i = 0; i < 8; ++i) { r[i] = lo[i]; r[i + 8] = hi[i]; }
    return r;
}

// C[M,N] (f32) = A[M,K] (f16, row-major) x W[N,K]^T (f16, row-major) + bias[N]
// M,N multiples of 128; K multiple of 32.
__global__ __launch_bounds__(256)
void gemm_f16_wmma(const _Float16* __restrict__ A,
                   const _Float16* __restrict__ W,
                   const float* __restrict__ bias,
                   float* __restrict__ C,
                   int M, int N, int K)
{
    __shared__ _Float16 As[2][BM * LDA];
    __shared__ _Float16 Bs[2][BN * LDB];

    const int tid   = threadIdx.x;
    const int lane  = tid & 31;
    const int wave  = tid >> 5;      // 0..7
    const int waveM = wave >> 2;     // 0..1 -> 64-row strip
    const int waveN = wave & 3;      // 0..3 -> 32-col strip
    const int h     = lane >> 4;     // lane half (K-group select)
    const int lr    = lane & 15;     // row/col within 16

    const int blockM = blockIdx.y * BM;
    const int blockN = blockIdx.x * BN;

    // Staging: 512 chunks of 16B per tile, 2 per thread: c in {tid, tid+256}
    const int r0  = tid >> 2;              // 0..63
    const int r1  = r0 + 64;               // 64..127
    const int kc  = (tid & 3) * 8;         // 0,8,16,24
    const _Float16* ag0 = A + (size_t)(blockM + r0) * K + kc;
    const _Float16* ag1 = A + (size_t)(blockM + r1) * K + kc;
    const _Float16* bg0 = W + (size_t)(blockN + r0) * K + kc;
    const _Float16* bg1 = W + (size_t)(blockN + r1) * K + kc;

    auto stage = [&](int k0, int buf) {
        async_copy16(ag0 + k0, &As[buf][r0 * LDA + kc]);
        async_copy16(ag1 + k0, &As[buf][r1 * LDA + kc]);
        async_copy16(bg0 + k0, &Bs[buf][r0 * LDB + kc]);
        async_copy16(bg1 + k0, &Bs[buf][r1 * LDB + kc]);
    };

    v8f acc[4][2];
    const v8f vzero = {};
#pragma unroll
    for (int i = 0; i < 4; ++i)
#pragma unroll
        for (int j = 0; j < 2; ++j) acc[i][j] = vzero;

    const int nK = K / BK;
    stage(0, 0);

    for (int kt = 0; kt < nK; ++kt) {
        const int cur  = kt & 1;
        const bool more = (kt + 1 < nK);
        if (more) stage((kt + 1) * BK, cur ^ 1);   // prefetch next tile behind compute
        wait_async_le(more ? 4 : 0);               // tile kt landed (4 = next tile in flight)
        __syncthreads();

        const _Float16* as = As[cur];
        const _Float16* bs = Bs[cur];

        // ---- fragments per ISA 16-bit A(16x32)/B(32x16) lane layouts ----
        v16h afrag[4];
#pragma unroll
        for (int fm = 0; fm < 4; ++fm) {
            int m = waveM * 64 + fm * 16 + lr;
            v8h lo = *reinterpret_cast<const v8h*>(&as[m * LDA + h * 8]);       // K = 8h..8h+7
            v8h hi = *reinterpret_cast<const v8h*>(&as[m * LDA + 16 + h * 8]);  // K = 16+8h..
            afrag[fm] = combine8(lo, hi);
        }
        v16h bfrag[2];
#pragma unroll
        for (int fn = 0; fn < 2; ++fn) {
            int n = waveN * 32 + fn * 16 + lr;
            v8h lo = *reinterpret_cast<const v8h*>(&bs[n * LDB + h * 8]);       // K = 8h..8h+7
            v8h hi = *reinterpret_cast<const v8h*>(&bs[n * LDB + 16 + h * 8]);
            bfrag[fn] = combine8(lo, hi);
        }

#pragma unroll
        for (int fm = 0; fm < 4; ++fm)
#pragma unroll
            for (int fn = 0; fn < 2; ++fn)
                acc[fm][fn] = __builtin_amdgcn_wmma_f32_16x16x32_f16(
                    false, afrag[fm], false, bfrag[fn],
                    (short)0, acc[fm][fn], false, false);
        __syncthreads();   // all waves done reading buf `cur` before it is re-staged
    }

    // ---- epilogue: C/D layout -> lane lr = col, half h picks M+8, VGPR j = M row ----
#pragma unroll
    for (int fm = 0; fm < 4; ++fm) {
#pragma unroll
        for (int fn = 0; fn < 2; ++fn) {
            int col  = blockN + waveN * 32 + fn * 16 + lr;
            float bv = bias[col];
#pragma unroll
            for (int j = 0; j < 8; ++j) {
                int row = blockM + waveM * 64 + fm * 16 + h * 8 + j;
                C[(size_t)row * N + col] = acc[fm][fn][j] + bv;
            }
        }
    }
}

// fp32 -> f16 conversion
__global__ void f32_to_f16_kernel(const float* __restrict__ src, _Float16* __restrict__ dst, int n) {
    int i = blockIdx.x * blockDim.x + threadIdx.x;
    if (i < n) dst[i] = (_Float16)src[i];
}

// Layer 1: static current -> spikes for all T (current identical every step)
__global__ void lif_static_kernel(const float* __restrict__ cur, _Float16* __restrict__ S, int n) {
    int i = blockIdx.x * blockDim.x + threadIdx.x;
    if (i >= n) return;
    float c = cur[i];
    float m = 0.0f;
#pragma unroll
    for (int t = 0; t < TSTEPS; ++t) {
        m = BETA * m + c;
        float s = (m > 1.0f) ? 1.0f : 0.0f;   // heaviside(m - threshold)
        m -= s;                                // subtract reset
        S[(size_t)t * n + i] = (_Float16)s;
    }
}

// Middle layer: per-timestep currents [T][n] -> spikes [T][n]
__global__ void lif_seq_kernel(const float* __restrict__ cur, _Float16* __restrict__ S, int n) {
    int i = blockIdx.x * blockDim.x + threadIdx.x;
    if (i >= n) return;
    float m = 0.0f;
#pragma unroll
    for (int t = 0; t < TSTEPS; ++t) {
        float c = cur[(size_t)t * n + i];
        m = BETA * m + c;
        float s = (m > 1.0f) ? 1.0f : 0.0f;
        m -= s;
        S[(size_t)t * n + i] = (_Float16)s;
    }
}

// Output layer: currents [T][n] -> sum of spikes over T
__global__ void lif_out_kernel(const float* __restrict__ cur, float* __restrict__ out, int n) {
    int i = blockIdx.x * blockDim.x + threadIdx.x;
    if (i >= n) return;
    float m = 0.0f, a = 0.0f;
#pragma unroll
    for (int t = 0; t < TSTEPS; ++t) {
        float c = cur[(size_t)t * n + i];
        m = BETA * m + c;
        float s = (m > 1.0f) ? 1.0f : 0.0f;
        m -= s;
        a += s;
    }
    out[i] = a;
}

extern "C" void kernel_launch(void* const* d_in, const int* in_sizes, int n_in,
                              void* d_out, int out_size, void* d_ws, size_t ws_size,
                              hipStream_t stream) {
    (void)in_sizes; (void)n_in; (void)out_size; (void)ws_size;
    const float* x  = (const float*)d_in[0];   // [512,1024]
    const float* W1 = (const float*)d_in[1];   // [2048,1024]
    const float* b1 = (const float*)d_in[2];   // [2048]
    const float* W2 = (const float*)d_in[3];   // [2048,2048]
    const float* b2 = (const float*)d_in[4];   // [2048]
    const float* W3 = (const float*)d_in[5];   // [512,2048]
    const float* b3 = (const float*)d_in[6];   // [512]
    float* out = (float*)d_out;                // [512,512]

    const int B = 512, N1 = 2048, N2 = 2048, N3 = 512, K1 = 1024;
    const int MT = TSTEPS * B;                 // 8192

    char* ws = (char*)d_ws;
    size_t off = 0;
    auto alloc = [&](size_t bytes) -> void* {
        void* p = ws + off;
        off += (bytes + 255) & ~(size_t)255;
        return p;
    };
    _Float16* Xh   = (_Float16*)alloc((size_t)B  * K1 * 2);   // 1 MB
    _Float16* W1h  = (_Float16*)alloc((size_t)N1 * K1 * 2);   // 4 MB
    _Float16* W2h  = (_Float16*)alloc((size_t)N2 * N1 * 2);   // 8 MB
    _Float16* W3h  = (_Float16*)alloc((size_t)N3 * N2 * 2);   // 2 MB
    float*    CUR1 = (float*)   alloc((size_t)B  * N1 * 4);   // 4 MB
    _Float16* S1   = (_Float16*)alloc((size_t)MT * N1 * 2);   // 32 MB
    float*    CUR2 = (float*)   alloc((size_t)MT * N2 * 4);   // 64 MB
    _Float16* S2   = (_Float16*)alloc((size_t)MT * N2 * 2);   // 32 MB
    float*    CUR3 = (float*)   alloc((size_t)MT * N3 * 4);   // 16 MB

    dim3 blk(256);
    auto g1 = [](int n) { return dim3((n + 255) / 256); };

    // fp32 -> f16 conversions
    f32_to_f16_kernel<<<g1(B * K1),  blk, 0, stream>>>(x,  Xh,  B * K1);
    f32_to_f16_kernel<<<g1(N1 * K1), blk, 0, stream>>>(W1, W1h, N1 * K1);
    f32_to_f16_kernel<<<g1(N2 * N1), blk, 0, stream>>>(W2, W2h, N2 * N1);
    f32_to_f16_kernel<<<g1(N3 * N2), blk, 0, stream>>>(W3, W3h, N3 * N2);

    // Layer 1 GEMM (once — x is static across timesteps): CUR1 = x @ W1^T + b1
    gemm_f16_wmma<<<dim3(N1 / BN, B / BM), blk, 0, stream>>>(Xh, W1h, b1, CUR1, B, N1, K1);
    // Layer 1 LIF: elementwise -> spikes for all 16 timesteps
    lif_static_kernel<<<g1(B * N1), blk, 0, stream>>>(CUR1, S1, B * N1);

    // Layer 2: one batched GEMM over all timesteps: CUR2[t] = S1[t] @ W2^T + b2
    gemm_f16_wmma<<<dim3(N2 / BN, MT / BM), blk, 0, stream>>>(S1, W2h, b2, CUR2, MT, N2, N1);
    lif_seq_kernel<<<g1(B * N2), blk, 0, stream>>>(CUR2, S2, B * N2);

    // Layer 3: batched GEMM: CUR3[t] = S2[t] @ W3^T + b3, then sum spikes over T
    gemm_f16_wmma<<<dim3(N3 / BN, MT / BM), blk, 0, stream>>>(S2, W3h, b3, CUR3, MT, N3, N2);
    lif_out_kernel<<<g1(B * N3), blk, 0, stream>>>(CUR3, out, B * N3);
}
